// ImplicitGeometricPriors_45698452029979
// MI455X (gfx1250) — matrix operand order repaired and checked by reading, not compiled, not run on hardware
//
#include <hip/hip_runtime.h>
#include <math.h>

// Problem dims (fixed by reference)
#define B_   2
#define NP_  512
#define D_   768
#define K_   64
#define H_   8
#define HD_  96
#define BN_  (B_*NP_)      // 1024 points total
#define CTXW (2*D_)        // 1536

typedef __bf16 bf16;
typedef __attribute__((ext_vector_type(16))) __bf16 v16bf;
typedef __attribute__((ext_vector_type(8)))  float  v8f;

// ---------------------------------------------------------------------------
// WMMA fragment loaders (CDNA5 16x16x32 bf16 layouts, CDNA5 ISA)
// A (16x32, row-major src, stride lda): lane m=lane&15 row, hi=lane>>4 K+16.
//   Two contiguous 8-element chunks -> 2x b128 loads.
// B (32x16): lane n=lane&15 col, hi selects K in [16hi,16hi+15].
//   With N-MAJOR source (Bt[n][k], stride ldk): 16 contiguous -> 2x b128.
// C/D f32 16x16: VGPR r -> row = r + 8*hi, col = lane&15
// ---------------------------------------------------------------------------
__device__ __forceinline__ v16bf frag_A(const bf16* A, int lda) {
  const int lane = threadIdx.x & 31;
  const int m = lane & 15, hi = lane >> 4;
  v16bf f;
#pragma unroll
  for (int j = 0; j < 8; ++j) {
    const int kb = 2*j + hi*16;
    f[2*j]   = A[(size_t)m*lda + kb];
    f[2*j+1] = A[(size_t)m*lda + kb + 1];
  }
  return f;
}

__device__ __forceinline__ v16bf frag_Bt(const bf16* Bt, int ldk) {
  const int lane = threadIdx.x & 31;
  const int n = lane & 15, hi = lane >> 4;
  const bf16* p = Bt + (size_t)n*ldk + hi*16;
  v16bf f;
#pragma unroll
  for (int i = 0; i < 16; ++i) f[i] = p[i];
  return f;
}

#define WMMA_BF16(a, b, c) \
  __builtin_amdgcn_wmma_f32_16x16x32_bf16(false, (a), false, (b), (short)0, (c), false, false)

// Single 16x16 tile, B N-major, software-pipelined (next frags loaded
// before current WMMA issues -> partial s_wait_loadcnt instead of drain)
__device__ __forceinline__ v8f gemm1(const bf16* A, int lda,
                                     const bf16* Bt, int ldk, int ksteps) {
  v8f acc = {};
  v16bf a = frag_A(A, lda);
  v16bf b = frag_Bt(Bt, ldk);
  for (int s = 0; s < ksteps - 1; ++s) {
    v16bf an = frag_A(A + (s+1)*32, lda);
    v16bf bn = frag_Bt(Bt + (s+1)*32, ldk);
    acc = WMMA_BF16(a, b, acc);
    a = an; b = bn;
  }
  return WMMA_BF16(a, b, acc);
}

// Four adjacent 16x16 N-tiles per wave, software-pipelined double buffer
struct Acc4 { v8f a0, a1, a2, a3; };
__device__ __forceinline__ Acc4 gemm4(const bf16* A, int lda,
                                      const bf16* Bt, int ldk, int ksteps) {
  Acc4 r; r.a0 = {}; r.a1 = {}; r.a2 = {}; r.a3 = {};
  v16bf a  = frag_A(A, lda);
  v16bf b0 = frag_Bt(Bt + (size_t)0*16*ldk, ldk);
  v16bf b1 = frag_Bt(Bt + (size_t)1*16*ldk, ldk);
  v16bf b2 = frag_Bt(Bt + (size_t)2*16*ldk, ldk);
  v16bf b3 = frag_Bt(Bt + (size_t)3*16*ldk, ldk);
  for (int s = 0; s < ksteps - 1; ++s) {
    __builtin_prefetch((const void*)(A + (s+2)*32), 0, 1);  // global_prefetch_b8
    const int o = (s+1)*32;
    v16bf an  = frag_A(A + o, lda);
    v16bf bn0 = frag_Bt(Bt + (size_t)0*16*ldk + o, ldk);
    v16bf bn1 = frag_Bt(Bt + (size_t)1*16*ldk + o, ldk);
    v16bf bn2 = frag_Bt(Bt + (size_t)2*16*ldk + o, ldk);
    v16bf bn3 = frag_Bt(Bt + (size_t)3*16*ldk + o, ldk);
    r.a0 = WMMA_BF16(a, b0, r.a0);
    r.a1 = WMMA_BF16(a, b1, r.a1);
    r.a2 = WMMA_BF16(a, b2, r.a2);
    r.a3 = WMMA_BF16(a, b3, r.a3);
    a = an; b0 = bn0; b1 = bn1; b2 = bn2; b3 = bn3;
  }
  r.a0 = WMMA_BF16(a, b0, r.a0);
  r.a1 = WMMA_BF16(a, b1, r.a1);
  r.a2 = WMMA_BF16(a, b2, r.a2);
  r.a3 = WMMA_BF16(a, b3, r.a3);
  return r;
}

// ---------------------------------------------------------------------------
// Converters (weights go to bf16; B-operands stored N-major = transposed)
// ---------------------------------------------------------------------------
__global__ void k_cvt(const float* __restrict__ src, bf16* __restrict__ dst, int n) {
  for (int i = blockIdx.x*blockDim.x + threadIdx.x; i < n; i += gridDim.x*blockDim.x)
    dst[i] = (bf16)src[i];
}
// WT[c][r] = W[r][c]; WT has leading dim = rows
__global__ void k_cvtT(const float* __restrict__ W, bf16* __restrict__ WT, int rows, int cols) {
  for (int i = blockIdx.x*blockDim.x + threadIdx.x; i < rows*cols; i += gridDim.x*blockDim.x) {
    int r = i / cols, c = i % cols;
    WT[(size_t)c*rows + r] = (bf16)W[(size_t)r*cols + c];
  }
}
__global__ void k_feat2ctx(const float* __restrict__ f, bf16* __restrict__ ctx) {
  for (int i = blockIdx.x*blockDim.x + threadIdx.x; i < BN_*D_; i += gridDim.x*blockDim.x) {
    int n = i / D_, d = i % D_;
    ctx[(size_t)n*CTXW + d] = (bf16)f[i];
  }
}

// ---------------------------------------------------------------------------
// Top-K nearest neighbors (one workgroup per point; 128 threads)
// ---------------------------------------------------------------------------
__global__ __launch_bounds__(128) void k_topk(const float* __restrict__ xyz,
                                              int* __restrict__ idx,
                                              float* __restrict__ nbrd) {
  __shared__ float dist[NP_];
  __shared__ float rv[128];
  __shared__ int   ri[128];
  const int n = blockIdx.x, b = n / NP_, p = n % NP_, t = threadIdx.x;
  const float px = xyz[((size_t)(b*NP_ + p))*3 + 0];
  const float py = xyz[((size_t)(b*NP_ + p))*3 + 1];
  const float pz = xyz[((size_t)(b*NP_ + p))*3 + 2];
  for (int j = t; j < NP_; j += 128) {
    float dx = px - xyz[((size_t)(b*NP_ + j))*3 + 0];
    float dy = py - xyz[((size_t)(b*NP_ + j))*3 + 1];
    float dz = pz - xyz[((size_t)(b*NP_ + j))*3 + 2];
    dist[j] = sqrtf(fmaxf(dx*dx + dy*dy + dz*dz, 0.f));
  }
  __syncthreads();
  for (int s = 0; s < K_; ++s) {
    float bv = 3.0e38f; int bi = NP_;
    for (int j = t; j < NP_; j += 128) {
      float v = dist[j];
      if (v < bv || (v == bv && j < bi)) { bv = v; bi = j; }
    }
    rv[t] = bv; ri[t] = bi;
    __syncthreads();
    for (int o = 64; o > 0; o >>= 1) {
      if (t < o) {
        if (rv[t+o] < rv[t] || (rv[t+o] == rv[t] && ri[t+o] < ri[t])) {
          rv[t] = rv[t+o]; ri[t] = ri[t+o];
        }
      }
      __syncthreads();
    }
    if (t == 0) {
      idx[(size_t)n*K_ + s]  = ri[0];
      nbrd[(size_t)n*K_ + s] = rv[0];
      dist[ri[0]] = 3.4e38f;
    }
    __syncthreads();
  }
}

// ---------------------------------------------------------------------------
// Q projection: q = ctx[:, :768] @ Wq + bq   (B = WqT, N-major)
// wave handles 4 N-tiles: grid = 64 rows * 12 groups / 8 waves = 96 blocks
// ---------------------------------------------------------------------------
__global__ __launch_bounds__(256) void k_qproj(const bf16* __restrict__ ctx,
                                               const bf16* __restrict__ WqT,
                                               const float* __restrict__ bq,
                                               bf16* __restrict__ qb) {
  const int wave = blockIdx.x*8 + (threadIdx.x >> 5);
  const int tm = wave / 12, tn0 = (wave % 12) * 4;
  if (tm >= BN_/16) return;
  Acc4 acc = gemm4(ctx + (size_t)tm*16*CTXW, CTXW,
                   WqT + (size_t)tn0*16*D_, D_, D_/32);
  const int lane = threadIdx.x & 31, hi = lane >> 4, nn = lane & 15;
  v8f* accs = &acc.a0;
#pragma unroll
  for (int tt = 0; tt < 4; ++tt) {
    const int col = (tn0 + tt)*16 + nn;
    const float bias = bq[col];
    v8f a = accs[tt];
#pragma unroll
    for (int r = 0; r < 8; ++r)
      qb[(size_t)(tm*16 + r + hi*8)*D_ + col] = (bf16)(a[r] + bias);
  }
}

// ---------------------------------------------------------------------------
// QK precompute (per head): qk[n,h,d] = (1/sqrt96) * sum_c q[n,h,c]*Wk[d,96h+c]
// B N-major is exactly Wk row-major (rows d, fast axis c) -> no transpose!
// ---------------------------------------------------------------------------
__global__ __launch_bounds__(256) void k_qk(const bf16* __restrict__ qb,
                                            const bf16* __restrict__ Wkb,
                                            bf16* __restrict__ qk) {
  const int h = blockIdx.y;
  const int wave = blockIdx.x*8 + (threadIdx.x >> 5);
  const int tm = wave / 12, tn0 = (wave % 12) * 4;
  if (tm >= BN_/16) return;
  Acc4 acc = gemm4(qb + (size_t)tm*16*D_ + h*HD_, D_,
                   Wkb + (size_t)tn0*16*D_ + h*HD_, D_, HD_/32);
  const float scale = 0.10206207261596575f; // 1/sqrt(96)
  const int lane = threadIdx.x & 31, hi = lane >> 4, nn = lane & 15;
  v8f* accs = &acc.a0;
#pragma unroll
  for (int tt = 0; tt < 4; ++tt) {
    const int col = (tn0 + tt)*16 + nn;
    v8f a = accs[tt];
#pragma unroll
    for (int r = 0; r < 8; ++r) {
      const int row = tm*16 + r + hi*8;
      qk[((size_t)row*H_ + h)*D_ + col] = (bf16)(a[r] * scale);
    }
  }
}

// ---------------------------------------------------------------------------
// Fused per-point attention core (one workgroup per point, 8 waves):
//   build kv[64][768] bf16 in LDS -> WMMA scores[8][64] -> softmax ->
//   WMMA wsum[8][768] -> global (bf16)
// ---------------------------------------------------------------------------
#define SM_KV   0
#define SM_SC   (SM_KV  + K_*D_*2)
#define SM_AT   (SM_SC  + H_*K_*4)
#define SM_WD   (SM_AT  + 16*K_*2)
#define SM_BD   (SM_WD  + D_*4)
#define SM_IDX  (SM_BD  + D_*4)
#define SM_DST  (SM_IDX + K_*4)
#define SM_TOT  (SM_DST + K_*4)

__global__ __launch_bounds__(256) void k_attn_core(
    const float* __restrict__ feat, const float* __restrict__ Wd,
    const float* __restrict__ bd, const int* __restrict__ idx,
    const float* __restrict__ nbrd, const bf16* __restrict__ qk,
    bf16* __restrict__ wsum) {
  extern __shared__ char smem[];
  bf16  (*kv)[D_]  = (bf16(*)[D_])(smem + SM_KV);
  float (*sc)[K_]  = (float(*)[K_])(smem + SM_SC);
  bf16  (*at)[K_]  = (bf16(*)[K_])(smem + SM_AT);
  float* swd  = (float*)(smem + SM_WD);
  float* sbd  = (float*)(smem + SM_BD);
  int*   sidx = (int*)(smem + SM_IDX);
  float* sdst = (float*)(smem + SM_DST);

  const int n = blockIdx.x, b = n / NP_, t = threadIdx.x;
  const int wave = t >> 5, lane = t & 31;

  if (t < K_) { sidx[t] = idx[(size_t)n*K_ + t]; sdst[t] = nbrd[(size_t)n*K_ + t]; }
  for (int d = t; d < D_; d += 256) { swd[d] = Wd[d]; sbd[d] = bd[d]; }
  __syncthreads();

  // kv[j][d] = feat[b, idx[j], d] + dist[j]*Wd[d] + bd[d]
  // vectorized: float4 gathers (b128) + 4x bf16 LDS stores (b64)
  {
    const int VD = D_/4;                       // 192 float4 per row
    for (int v = t; v < K_*VD; v += 256) {
      const int j = v / VD, d4 = (v - j*VD)*4;
      const float4 f4 = *(const float4*)&feat[((size_t)(b*NP_ + sidx[j]))*D_ + d4];
      const float dj = sdst[j];
      bf16* dst = &kv[j][d4];
      dst[0] = (bf16)(f4.x + dj*swd[d4+0] + sbd[d4+0]);
      dst[1] = (bf16)(f4.y + dj*swd[d4+1] + sbd[d4+1]);
      dst[2] = (bf16)(f4.z + dj*swd[d4+2] + sbd[d4+2]);
      dst[3] = (bf16)(f4.w + dj*swd[d4+3] + sbd[d4+3]);
    }
  }
  __syncthreads();

  // scores[h][j] = sum_d qk[n,h,d] * kv[j][d]
  // B operand is kv^T, and row-major kv IS its N-major form -> b128 ds loads
  if (wave < 4) {
    v8f acc = {};
    const bf16* qkn = qk + (size_t)n*H_*D_;
    const int m = lane & 15, hi = lane >> 4;
    const int h = m & 7;                    // rows 0..7 = heads, 8..15 dup
    for (int s = 0; s < D_/32; ++s) {
      v16bf a;
#pragma unroll
      for (int j = 0; j < 8; ++j) {
        const int kb = 2*j + hi*16 + s*32;
        a[2*j]   = qkn[(size_t)h*D_ + kb];
        a[2*j+1] = qkn[(size_t)h*D_ + kb + 1];
      }
      v16bf bb = frag_Bt(&kv[wave*16][0] + s*32, D_);
      acc = WMMA_BF16(a, bb, acc);
    }
    if (lane < 16) {
      const int col = wave*16 + lane;
#pragma unroll
      for (int r = 0; r < 8; ++r) sc[r][col] = acc[r];
    }
  }
  __syncthreads();

  // softmax over K=64 per head (wave h owns head h; 2 values per lane)
  if (wave < H_) {
    const int h = wave;
    float s0 = sc[h][lane], s1 = sc[h][lane + 32];
    float m = fmaxf(s0, s1);
    for (int o = 16; o > 0; o >>= 1) m = fmaxf(m, __shfl_xor(m, o));
    float e0 = __expf(s0 - m), e1 = __expf(s1 - m);
    float sum = e0 + e1;
    for (int o = 16; o > 0; o >>= 1) sum += __shfl_xor(sum, o);
    const float inv = 1.f / sum;
    at[h][lane]        = (bf16)(e0 * inv);
    at[h][lane + 32]   = (bf16)(e1 * inv);
    at[h+8][lane]      = (bf16)0.f;      // zero pad rows 8..15
    at[h+8][lane + 32] = (bf16)0.f;
  }
  __syncthreads();

  // wsum[h][d] = sum_j attn[h][j] * kv[j][d]  (M=16 padded, N=768, K=64)
  const int hi = lane >> 4, ncol = lane & 15;
  for (int tt = wave; tt < D_/16; tt += 8) {
    v8f acc = {};
#pragma unroll
    for (int s = 0; s < K_/32; ++s) {
      v16bf a = frag_A(&at[0][0] + s*32, K_);
      v16bf bb;
#pragma unroll
      for (int j = 0; j < 8; ++j) {
        const int k = s*32 + hi*16 + 2*j;   // k = neighbor index
        bb[2*j]   = kv[k][tt*16 + ncol];
        bb[2*j+1] = kv[k+1][tt*16 + ncol];
      }
      acc = WMMA_BF16(a, bb, acc);
    }
    if (hi == 0) {     // rows 0..7 valid (heads)
      const int col = tt*16 + ncol;
#pragma unroll
      for (int r = 0; r < 8; ++r)
        wsum[((size_t)n*H_ + r)*D_ + col] = (bf16)acc[r];
    }
  }
}

// ---------------------------------------------------------------------------
// Per-head O projection: o[n,96h+c] = sum_d wsum[n,h,d]*Wv[d,96h+c] + bv
// B = WvT (N-major): rows c' = output col, fast axis d
// ---------------------------------------------------------------------------
__global__ __launch_bounds__(256) void k_oproj(const bf16* __restrict__ wsum,
                                               const bf16* __restrict__ WvT,
                                               const float* __restrict__ bv,
                                               bf16* __restrict__ ob) {
  const int h = blockIdx.y;
  const int wave = blockIdx.x*8 + (threadIdx.x >> 5);
  const int TN = HD_/16;  // 6
  const int tm = wave / TN, tn = wave % TN;
  if (tm >= BN_/16) return;
  v8f acc = gemm1(wsum + (size_t)tm*16*(H_*D_) + (size_t)h*D_, H_*D_,
                  WvT + (size_t)(h*HD_ + tn*16)*D_, D_, D_/32);
  const int lane = threadIdx.x & 31, hi = lane >> 4, c = tn*16 + (lane & 15);
  const float bias = bv[h*HD_ + c];
#pragma unroll
  for (int r = 0; r < 8; ++r)
    ob[(size_t)(tm*16 + r + hi*8)*D_ + h*HD_ + c] = (bf16)(acc[r] + bias);
}

// ---------------------------------------------------------------------------
// attended = o @ Wo + bo -> ctx[:, 768:1536]   (B = WoT, N-major)
// ---------------------------------------------------------------------------
__global__ __launch_bounds__(256) void k_attn_out(const bf16* __restrict__ ob,
                                                  const bf16* __restrict__ WoT,
                                                  const float* __restrict__ bo,
                                                  bf16* __restrict__ ctx) {
  const int wave = blockIdx.x*8 + (threadIdx.x >> 5);
  const int tm = wave / 12, tn0 = (wave % 12) * 4;
  if (tm >= BN_/16) return;
  Acc4 acc = gemm4(ob + (size_t)tm*16*D_, D_,
                   WoT + (size_t)tn0*16*D_, D_, D_/32);
  const int lane = threadIdx.x & 31, hi = lane >> 4, nn = lane & 15;
  v8f* accs = &acc.a0;
#pragma unroll
  for (int tt = 0; tt < 4; ++tt) {
    const int col = (tn0 + tt)*16 + nn;
    const float bias = bo[col];
    v8f a = accs[tt];
#pragma unroll
    for (int r = 0; r < 8; ++r)
      ctx[(size_t)(tm*16 + r + hi*8)*CTXW + D_ + col] = (bf16)(a[r] + bias);
  }
}

// ---------------------------------------------------------------------------
// Enhancer GEMM: y = ctx @ We + be (fp32 out, K=1536)  (B = WeT, N-major)
// ---------------------------------------------------------------------------
__global__ __launch_bounds__(256) void k_enhance(const bf16* __restrict__ ctx,
                                                 const bf16* __restrict__ WeT,
                                                 const float* __restrict__ be,
                                                 float* __restrict__ y) {
  const int wave = blockIdx.x*8 + (threadIdx.x >> 5);
  const int tm = wave / 12, tn0 = (wave % 12) * 4;
  if (tm >= BN_/16) return;
  Acc4 acc = gemm4(ctx + (size_t)tm*16*CTXW, CTXW,
                   WeT + (size_t)tn0*16*CTXW, CTXW, CTXW/32);
  const int lane = threadIdx.x & 31, hi = lane >> 4, nn = lane & 15;
  v8f* accs = &acc.a0;
#pragma unroll
  for (int tt = 0; tt < 4; ++tt) {
    const int col = (tn0 + tt)*16 + nn;
    const float bias = be[col];
    v8f a = accs[tt];
#pragma unroll
    for (int r = 0; r < 8; ++r)
      y[(size_t)(tm*16 + r + hi*8)*D_ + col] = a[r] + bias;
  }
}

// ---------------------------------------------------------------------------
// LayerNorm + SiLU per row (768), one workgroup per row
// ---------------------------------------------------------------------------
__global__ __launch_bounds__(256) void k_ln_silu(const float* __restrict__ y,
                                                 const float* __restrict__ g,
                                                 const float* __restrict__ b,
                                                 float* __restrict__ out) {
  __shared__ float red[256];
  const int row = blockIdx.x, t = threadIdx.x;
  const float* yr = y + (size_t)row*D_;
  const float l0 = yr[t], l1 = yr[t+256], l2 = yr[t+512];
  red[t] = l0 + l1 + l2;
  __syncthreads();
  for (int o = 128; o > 0; o >>= 1) { if (t < o) red[t] += red[t+o]; __syncthreads(); }
  const float mean = red[0] * (1.f/D_);
  __syncthreads();
  const float d0 = l0-mean, d1 = l1-mean, d2 = l2-mean;
  red[t] = d0*d0 + d1*d1 + d2*d2;
  __syncthreads();
  for (int o = 128; o > 0; o >>= 1) { if (t < o) red[t] += red[t+o]; __syncthreads(); }
  const float inv = rsqrtf(red[0] * (1.f/D_) + 1e-5f);
  float* orow = out + (size_t)row*D_;
  const float x0 = d0*inv*g[t]     + b[t];
  const float x1 = d1*inv*g[t+256] + b[t+256];
  const float x2 = d2*inv*g[t+512] + b[t+512];
  orow[t]     = x0 / (1.f + __expf(-x0));
  orow[t+256] = x1 / (1.f + __expf(-x1));
  orow[t+512] = x2 / (1.f + __expf(-x2));
}

// ---------------------------------------------------------------------------
extern "C" void kernel_launch(void* const* d_in, const int* in_sizes, int n_in,
                              void* d_out, int out_size, void* d_ws, size_t ws_size,
                              hipStream_t stream) {
  const float* feat = (const float*)d_in[0];
  const float* xyz  = (const float*)d_in[1];
  const float* Wd   = (const float*)d_in[2];
  const float* bd   = (const float*)d_in[3];
  const float* Wq   = (const float*)d_in[4];
  const float* bq   = (const float*)d_in[5];
  const float* Wk   = (const float*)d_in[6];
  // d_in[7] = bk: constant per head across neighbors -> cancels in softmax
  const float* Wv   = (const float*)d_in[8];
  const float* bv   = (const float*)d_in[9];
  const float* Wo   = (const float*)d_in[10];
  const float* bo   = (const float*)d_in[11];
  const float* We   = (const float*)d_in[12];
  const float* be   = (const float*)d_in[13];
  const float* gln  = (const float*)d_in[14];
  const float* bln  = (const float*)d_in[15];
  float* out = (float*)d_out;

  char* ws = (char*)d_ws;
  size_t off = 0;
  auto alloc = [&](size_t bytes) -> void* {
    void* p = ws + off;
    off = (off + bytes + 255) & ~(size_t)255;
    return p;
  };
  int*   idx  = (int*)  alloc((size_t)BN_*K_*4);
  float* nbrd = (float*)alloc((size_t)BN_*K_*4);
  bf16*  ctxb = (bf16*) alloc((size_t)BN_*CTXW*2);
  bf16*  WqT  = (bf16*) alloc((size_t)D_*D_*2);
  bf16*  Wkb  = (bf16*) alloc((size_t)D_*D_*2);
  bf16*  WvT  = (bf16*) alloc((size_t)D_*D_*2);
  bf16*  WoT  = (bf16*) alloc((size_t)D_*D_*2);
  bf16*  WeT  = (bf16*) alloc((size_t)CTXW*D_*2);
  bf16*  qb   = (bf16*) alloc((size_t)BN_*D_*2);
  bf16*  qkb  = (bf16*) alloc((size_t)BN_*H_*D_*2);
  bf16*  wsb  = (bf16*) alloc((size_t)BN_*H_*D_*2);
  bf16*  obuf = (bf16*) alloc((size_t)BN_*D_*2);
  float* ybuf = (float*)alloc((size_t)BN_*D_*4);

  const int CW = D_*D_;
  k_cvtT<<<(CW+255)/256, 256, 0, stream>>>(Wq, WqT, D_, D_);
  k_cvt <<<(CW+255)/256, 256, 0, stream>>>(Wk, Wkb, CW);        // already N-major for QK
  k_cvtT<<<(CW+255)/256, 256, 0, stream>>>(Wv, WvT, D_, D_);
  k_cvtT<<<(CW+255)/256, 256, 0, stream>>>(Wo, WoT, D_, D_);
  k_cvtT<<<(CTXW*D_+255)/256, 256, 0, stream>>>(We, WeT, CTXW, D_);
  k_feat2ctx<<<(BN_*D_+255)/256, 256, 0, stream>>>(feat, ctxb);

  k_topk<<<BN_, 128, 0, stream>>>(xyz, idx, nbrd);

  // quad-tile GEMMs: 64 row-tiles * 12 groups of 4 N-tiles / 8 waves = 96 blocks
  k_qproj<<<96, 256, 0, stream>>>(ctxb, WqT, bq, qb);
  k_qk<<<dim3(96, H_), 256, 0, stream>>>(qb, Wkb, qkb);

  k_attn_core<<<BN_, 256, SM_TOT, stream>>>(feat, Wd, bd, idx, nbrd, qkb, wsb);

  const int nTilesHead = (BN_/16) * (HD_/16);         // 384
  k_oproj<<<dim3(nTilesHead/8, H_), 256, 0, stream>>>(wsb, WvT, bv, obuf);

  k_attn_out<<<96, 256, 0, stream>>>(obuf, WoT, bo, ctxb);
  k_enhance<<<96, 256, 0, stream>>>(ctxb, WeT, be, ybuf);
  k_ln_silu<<<BN_, 256, 0, stream>>>(ybuf, gln, bln, out);
}
